// FrameTransformerEncoder_68917045231702
// MI455X (gfx1250) — compile-verified, matrix operand order checked by reading, not run on hardware
//
#include <hip/hip_runtime.h>
#include <hip/hip_bf16.h>

#define DI __device__ __forceinline__

typedef __bf16 bf16_t;
typedef __attribute__((ext_vector_type(8)))  bf16_t v8bf;
typedef __attribute__((ext_vector_type(16))) bf16_t v16bf;
typedef __attribute__((ext_vector_type(8)))  float  v8f;
typedef __attribute__((ext_vector_type(4)))  float  fvec4;
typedef __attribute__((ext_vector_type(4)))  unsigned v4u;
typedef __attribute__((ext_vector_type(8)))  unsigned v8u;

// ---- problem constants ----
constexpr int B_  = 2;
constexpr int C_  = 4;
constexpr int L_  = 1024;
constexpr int F_  = 256;   // CH*CW
constexpr int H_  = 8;
constexpr int D_  = 32;    // F/H
constexpr int EXP_ = 4;
constexpr int N_  = B_ * L_;        // 2048 samples
constexpr int BCH_ = B_ * C_ * H_;  // 64 attention batches
constexpr float EPS_ = 1e-5f;

// ---- helpers ----
DI bf16_t f2bf(float f) {
  unsigned u = __float_as_uint(f);
  u += 0x7FFFu + ((u >> 16) & 1u);           // round-to-nearest-even
  unsigned short s = (unsigned short)(u >> 16);
  bf16_t r;
  __builtin_memcpy(&r, &s, sizeof(r));
  return r;
}

// 16x32 bf16 fragment for one lane: elements 0..7 from `lo`, 8..15 from `hi`
// (the two contiguous 8-half K-blocks per the CDNA5 WMMA VGPR layout).
DI v16bf load_frag(const bf16_t* lo, const bf16_t* hi) {
  v8bf a = *(const v8bf*)lo;
  v8bf b = *(const v8bf*)hi;
  return __builtin_shufflevector(a, b, 0,1,2,3,4,5,6,7,8,9,10,11,12,13,14,15);
}

DI v8f wmma_bf16(v16bf a, v16bf b, v8f c) {
  return __builtin_amdgcn_wmma_f32_16x16x32_bf16(false, a, false, b,
                                                 (short)0, c, false, false);
}

DI float block_reduce_sum(float v, float* red) {
  int t = threadIdx.x;
  red[t] = v; __syncthreads();
  for (int s = 128; s > 0; s >>= 1) {
    if (t < s) red[t] += red[t + s];
    __syncthreads();
  }
  float r = red[0];
  __syncthreads();
  return r;
}

// Async global->LDS copy of one 16B chunk (ASYNCcnt-tracked).
DI void async_copy_b128(unsigned lds_byte_off, const void* gaddr) {
  asm volatile("global_load_async_to_lds_b128 %0, %1, off"
               :: "v"(lds_byte_off), "v"((unsigned long long)gaddr)
               : "memory");
}
DI void wait_async0() {
  asm volatile("s_wait_asynccnt 0x0" ::: "memory");
}

// Tensor Data Mover: DMA a 2D tile (rows x row_bytes, global stride row_bytes)
// into LDS with (pad_amount+1) DWORDs of LDS padding inserted after every
// 2^(pad_interval+1) DWORDs. One instruction, TENSORcnt-tracked, EXEC-ignored.
// D# groups built in SGPR tuples per CDNA5 ISA (2-group
// form: VADDR2/3 omitted == NULL, valid for tensors up to 2D).
DI void tdm_load_2d(unsigned lds_byte_off, const void* gaddr,
                    unsigned row_bytes, unsigned rows,
                    unsigned pad_interval, unsigned pad_amount) {
  unsigned long long ga = (unsigned long long)gaddr;
  v4u g0;
  g0[0] = 1u;                                        // count=1 (valid user D#)
  g0[1] = lds_byte_off;                              // lds_addr
  g0[2] = (unsigned)(ga & 0xFFFFFFFFull);            // global_addr[31:0]
  g0[3] = (unsigned)((ga >> 32) & 0x01FFFFFFull)     // global_addr[56:32]
        | (2u << 30);                                // type=2 ("image")
  v8u g1;
  g1[0] = (1u << 20) | (pad_interval << 22) | (pad_amount << 25); // pad_enable
  g1[1] = (row_bytes & 0xFFFFu) << 16;               // tensor_dim0[15:0]
  g1[2] = (row_bytes >> 16) | ((rows & 0xFFFFu) << 16); // dim0 hi / tensor_dim1 lo
  g1[3] = (rows >> 16) | ((row_bytes & 0xFFFFu) << 16); // dim1 hi / tile_dim0
  g1[4] = rows & 0xFFFFu;                            // tile_dim1 (tile_dim2=0)
  g1[5] = row_bytes;                                 // tensor_dim0_stride[31:0]
  g1[6] = 0u;                                        // stride hi / dim1_stride lo
  g1[7] = 0u;
  asm volatile("tensor_load_to_lds %0, %1" :: "s"(g0), "s"(g1) : "memory");
}

// Shared per-channel GEMM core: 16x16 tile, K=256 -> 8 WMMAs.
// Y[n(M), g(Ncol)] = sum_f X[n,c,f] * W[c,g,f]
DI v8f pw_core(const bf16_t* __restrict__ X, const bf16_t* __restrict__ W,
               int c, int mt, int nt, int lane) {
  int mrow = lane & 15, hsel = lane >> 4;
  const bf16_t* arow = X + ((size_t)(mt*16 + mrow) * C_ + c) * F_;
  const bf16_t* brow = W + (size_t)c * F_ * F_ + (size_t)(nt*16 + mrow) * F_;
  v8f acc = {0.f,0.f,0.f,0.f,0.f,0.f,0.f,0.f};
  #pragma unroll
  for (int kb = 0; kb < F_; kb += 32) {
    v16bf a = load_frag(arow + kb + hsel*8, arow + kb + 16 + hsel*8);
    v16bf b = load_frag(brow + kb + hsel*8, brow + kb + 16 + hsel*8);
    acc = wmma_bf16(a, b, acc);
  }
  return acc;
}

// ---- kernel 0: f32 -> bf16 conversion ----
__global__ void k_f32_to_bf16(const float* __restrict__ src, bf16_t* __restrict__ dst, int n) {
  int i = blockIdx.x * blockDim.x + threadIdx.x;
  if (i < n) dst[i] = f2bf(src[i]);
}

// ---- kernel 1: transpose + instance-norm + 3-tap conv (CW==1) ----
__global__ void k_prep(const float* __restrict__ x,
                       const float* __restrict__ g1, const float* __restrict__ b1,
                       const float* __restrict__ wqc, const float* __restrict__ wkc,
                       const float* __restrict__ wvc,
                       bf16_t* __restrict__ cq, bf16_t* __restrict__ ck,
                       bf16_t* __restrict__ cv) {
  __shared__ float nxs[4][258];   // +1 halo each side (zero pad)
  __shared__ float red[256];
  __shared__ float wc[3][48];     // center-column taps w[co][ci][kh][1]
  int n = blockIdx.x, t = threadIdx.x;
  int b = n >> 10, l = n & (L_ - 1);
  if (t < 48) { wc[0][t] = wqc[t*3+1]; wc[1][t] = wkc[t*3+1]; wc[2][t] = wvc[t*3+1]; }
  float xv[4];
  #pragma unroll
  for (int ci = 0; ci < 4; ++ci)
    xv[ci] = x[((size_t)(b*C_+ci)*L_ + l)*F_ + t];
  #pragma unroll
  for (int ci = 0; ci < 4; ++ci) {
    float s  = block_reduce_sum(xv[ci], red);
    float m  = s * (1.f/256.f);
    float s2 = block_reduce_sum(xv[ci]*xv[ci], red);
    float var = s2 * (1.f/256.f) - m*m;
    nxs[ci][t+1] = (xv[ci]-m) * rsqrtf(var + EPS_) * g1[ci] + b1[ci];
  }
  if (t < 4) { nxs[t][0] = 0.f; nxs[t][257] = 0.f; }
  __syncthreads();
  #pragma unroll
  for (int co = 0; co < 4; ++co) {
    float aq = 0.f, ak = 0.f, av = 0.f;
    #pragma unroll
    for (int ci = 0; ci < 4; ++ci)
      #pragma unroll
      for (int kh = 0; kh < 3; ++kh) {
        float xn = nxs[ci][t + kh];
        int wi = (co*4+ci)*3 + kh;
        aq += xn * wc[0][wi]; ak += xn * wc[1][wi]; av += xn * wc[2][wi];
      }
    size_t oi = ((size_t)n * C_ + co) * F_ + t;
    cq[oi] = f2bf(aq); ck[oi] = f2bf(ak); cv[oi] = f2bf(av);
  }
}

// ---- kernel 2a: pointwise GEMM, plain f32 output (o-projection) ----
__global__ void k_pw_gemm(const bf16_t* __restrict__ X, const bf16_t* __restrict__ W,
                          float* __restrict__ Y) {
  int wid  = (blockIdx.x * blockDim.x + threadIdx.x) >> 5;
  int lane = threadIdx.x & 31;
  int c  = wid >> 11;
  int mt = (wid >> 4) & 127;
  int nt = wid & 15;
  v8f acc = pw_core(X, W, c, mt, nt, lane);
  int col = lane & 15, rsel = lane >> 4;
  #pragma unroll
  for (int r = 0; r < 8; ++r) {
    int row = mt*16 + rsel*8 + r;
    Y[((size_t)row * C_ + c) * F_ + nt*16 + col] = acc[r];
  }
}

// ---- kernel 2b: pointwise GEMM fused with rotary, bf16 (bch,L,D) output ----
// Rotary pairs are adjacent output columns == adjacent lanes: one shfl_xor(1).
__global__ void k_pw_gemm_rot(const bf16_t* __restrict__ X, const bf16_t* __restrict__ W,
                              bf16_t* __restrict__ dst) {
  int wid  = (blockIdx.x * blockDim.x + threadIdx.x) >> 5;
  int lane = threadIdx.x & 31;
  int c  = wid >> 11;
  int mt = (wid >> 4) & 127;
  int nt = wid & 15;
  v8f acc = pw_core(X, W, c, mt, nt, lane);
  int col = lane & 15, rsel = lane >> 4;
  int f = nt*16 + col;
  int h = f >> 5, d = f & 31, pr = d >> 1;
  float invf = __expf(-(float)(2*pr) * (9.210340371976184f / 32.f)); // 10000^(-2pr/32)
  #pragma unroll
  for (int r = 0; r < 8; ++r) {
    int n = mt*16 + rsel*8 + r;
    int l = n & (L_ - 1), b = n >> 10;
    float ang = (float)l * invf;
    float cs = __cosf(ang), sn = __sinf(ang);
    float mine  = acc[r];
    float other = __shfl_xor(mine, 1, 32);
    float rot = (f & 1) ? (mine * cs + other * sn)   // odd slot: x2*cos + x1*sin
                        : (mine * cs - other * sn);  // even slot: x1*cos - x2*sin
    int bch = (b*C_ + c)*H_ + h;
    dst[((size_t)bch * L_ + l) * D_ + d] = f2bf(rot);
  }
}

// ---- kernel 2c: pointwise GEMM, bf16 transposed (bch,D,L) output for V ----
__global__ void k_pw_gemm_vT(const bf16_t* __restrict__ X, const bf16_t* __restrict__ W,
                             bf16_t* __restrict__ vhT) {
  int wid  = (blockIdx.x * blockDim.x + threadIdx.x) >> 5;
  int lane = threadIdx.x & 31;
  int c  = wid >> 11;
  int mt = (wid >> 4) & 127;
  int nt = wid & 15;
  v8f acc = pw_core(X, W, c, mt, nt, lane);
  int col = lane & 15, rsel = lane >> 4;
  int f = nt*16 + col;
  int h = f >> 5, d = f & 31;
  #pragma unroll
  for (int r = 0; r < 8; ++r) {
    int n = mt*16 + rsel*8 + r;
    int l = n & (L_ - 1), b = n >> 10;
    int bch = (b*C_ + c)*H_ + h;
    vhT[((size_t)bch * D_ + d) * L_ + l] = f2bf(acc[r]);
  }
}

// ---- kernel 3: QK^T/16 + prev_qk -> qk, fused online softmax stats ----
// One wave owns a 16-row strip and loops all 64 key tiles (K=D=32: one WMMA
// per tile). K panel staged to LDS by the Tensor Data Mover in two 512-row
// halves; TDM's pad feature inserts 16B after every 64B row -> 80B stride,
// conflict-free ds_load_b128. Row (max,sum) kept online per row-slot and
// merged across the 16 column lanes via shfl_xor butterflies.
__global__ void __launch_bounds__(256) k_qk_fused(
    const bf16_t* __restrict__ qh, const bf16_t* __restrict__ kh,
    const float* __restrict__ prevqk, float* __restrict__ qkout,
    float* __restrict__ rmax, float* __restrict__ rsum) {
  __shared__ bf16_t kpanel[512 * 40];   // 40960 bytes
  int blk = blockIdx.x;                 // 512 blocks: (bch, lt-group)
  int bch = blk >> 3;
  int ltg = blk & 7;
  int t = threadIdx.x;
  int wv = t >> 5, lane = t & 31;
  int lt = ltg * 8 + wv;
  int mrow = lane & 15, hsel = lane >> 4;

  const bf16_t* arow = qh + ((size_t)bch * L_ + lt*16 + mrow) * D_;
  v16bf a = load_frag(arow + hsel*8, arow + 16 + hsel*8);

  float m8[8], s8[8];
  #pragma unroll
  for (int r = 0; r < 8; ++r) { m8[r] = -1e30f; s8[r] = 0.f; }

  unsigned pbase = (unsigned)(size_t)(&kpanel[0]);
  const bf16_t* kbase = kh + (size_t)bch * L_ * D_;
  int col = lane & 15, rsel = lane >> 4;

  for (int half = 0; half < 2; ++half) {
    if (wv == 0) {    // wave-uniform: one TDM issue per block per half
      // 512 rows x 64B, global stride 64B; LDS pad 16B (4 DW, code 3) after
      // every 64B (16 DW, code 3)  ->  80B padded rows.
      tdm_load_2d(pbase, kbase + ((size_t)(half*512)) * D_,
                  /*row_bytes=*/64u, /*rows=*/512u,
                  /*pad_interval=*/3u, /*pad_amount=*/3u);
      __builtin_amdgcn_s_wait_tensorcnt(0);
    }
    __syncthreads();

    for (int mtl = 0; mtl < 32; ++mtl) {
      int mt = half * 32 + mtl;
      const bf16_t* brow = &kpanel[(mtl*16 + mrow) * 40];
      v16bf b = load_frag(brow + hsel*8, brow + 16 + hsel*8);
      v8f acc = {0.f,0.f,0.f,0.f,0.f,0.f,0.f,0.f};
      acc = wmma_bf16(a, b, acc);
      #pragma unroll
      for (int r = 0; r < 8; ++r) {
        size_t qi = ((size_t)bch*L_ + lt*16 + rsel*8 + r) * L_ + mt*16 + col;
        float val = acc[r] * 0.0625f + prevqk[qi];   // 1/sqrt(CH) = 1/16
        qkout[qi] = val;
        float mo = m8[r], mn = fmaxf(mo, val);
        s8[r] = s8[r] * __expf(mo - mn) + __expf(val - mn);
        m8[r] = mn;
      }
    }
    __syncthreads();   // all waves done reading before restage
  }
  // merge stats across the 16 column lanes (stays within each 16-lane half)
  #pragma unroll
  for (int x = 1; x < 16; x <<= 1) {
    #pragma unroll
    for (int r = 0; r < 8; ++r) {
      float mo = __shfl_xor(m8[r], x, 32);
      float so = __shfl_xor(s8[r], x, 32);
      float mn = fmaxf(m8[r], mo);
      s8[r] = s8[r] * __expf(m8[r] - mn) + so * __expf(mo - mn);
      m8[r] = mn;
    }
  }
  if ((lane & 15) == 0) {
    #pragma unroll
    for (int r = 0; r < 8; ++r) {
      int row = lt*16 + rsel*8 + r;
      rmax[bch * L_ + row] = m8[r];
      rsum[bch * L_ + row] = s8[r];
    }
  }
}

// ---- kernel 4: PV = softmax(qk) @ V, probabilities built on the fly ----
// V^T panel (16 x 1024) async-staged to LDS once per block (rows padded
// 1024->1032 halfs: conflict-free), dedicating the VMEM pipe to the qk stream.
__global__ void __launch_bounds__(256) k_av(
    const float* __restrict__ qk, const bf16_t* __restrict__ vhT,
    const float* __restrict__ rmax, const float* __restrict__ rsum,
    bf16_t* __restrict__ af) {
  __shared__ bf16_t vpanel[16 * 1032];   // 33024 bytes
  int blk = blockIdx.x;                  // 1024 blocks: (bch, nt, lt-group)
  int bch = blk >> 4;
  int nt  = (blk >> 3) & 1;
  int ltg = blk & 7;
  int t = threadIdx.x;
  int wv = t >> 5, lane = t & 31;
  int lt = ltg * 8 + wv;
  int mrow = lane & 15, hsel = lane >> 4;

  unsigned pbase = (unsigned)(size_t)(&vpanel[0]);
  const bf16_t* gbase = vhT + ((size_t)bch * D_ + nt*16) * L_;
  #pragma unroll
  for (int i = 0; i < 8; ++i) {
    int idx = i * 256 + t;                 // 16 rows x 128 chunks of 16B
    int row = idx >> 7, cch = idx & 127;
    async_copy_b128(pbase + (unsigned)(row * 2064 + cch * 16),
                    gbase + (size_t)row * L_ + cch*8);
  }
  wait_async0();
  __syncthreads();

  int l0 = lt * 16;
  const float* srow = qk + ((size_t)bch * L_ + l0 + mrow) * L_;
  float rm = rmax[bch * L_ + l0 + mrow];
  v8f acc = {0.f,0.f,0.f,0.f,0.f,0.f,0.f,0.f};
  for (int kb = 0; kb < L_; kb += 32) {
    const float* ap = srow + kb + hsel*8;
    __builtin_prefetch(ap + 32, 0, 1);
    fvec4 f0 = *(const fvec4*)(ap);
    fvec4 f1 = *(const fvec4*)(ap + 4);
    fvec4 f2 = *(const fvec4*)(ap + 16);
    fvec4 f3 = *(const fvec4*)(ap + 20);
    v16bf a;
    a[0]  = f2bf(__expf(f0.x - rm)); a[1]  = f2bf(__expf(f0.y - rm));
    a[2]  = f2bf(__expf(f0.z - rm)); a[3]  = f2bf(__expf(f0.w - rm));
    a[4]  = f2bf(__expf(f1.x - rm)); a[5]  = f2bf(__expf(f1.y - rm));
    a[6]  = f2bf(__expf(f1.z - rm)); a[7]  = f2bf(__expf(f1.w - rm));
    a[8]  = f2bf(__expf(f2.x - rm)); a[9]  = f2bf(__expf(f2.y - rm));
    a[10] = f2bf(__expf(f2.z - rm)); a[11] = f2bf(__expf(f2.w - rm));
    a[12] = f2bf(__expf(f3.x - rm)); a[13] = f2bf(__expf(f3.y - rm));
    a[14] = f2bf(__expf(f3.z - rm)); a[15] = f2bf(__expf(f3.w - rm));
    const bf16_t* brow = &vpanel[mrow * 1032 + kb + hsel*8];
    v16bf b = load_frag(brow, brow + 16);
    acc = wmma_bf16(a, b, acc);
  }
  int col = lane & 15, rsel = lane >> 4;
  int bb = bch >> 5, cc = (bch >> 3) & 3, hh = bch & 7;
  #pragma unroll
  for (int r = 0; r < 8; ++r) {
    int lrow = l0 + rsel*8 + r;
    float inv = 1.0f / rsum[bch * L_ + lrow];
    size_t oi = (((size_t)(bb * L_ + lrow)) * C_ + cc) * F_ + hh * D_ + nt*16 + col;
    af[oi] = f2bf(acc[r] * inv);
  }
}

// ---- kernel 5: concat + inorm + MLP bottleneck + dw-mix + residual ----
__global__ void k_mlp(const float* __restrict__ x, const float* __restrict__ o,
                      const float* __restrict__ g2, const float* __restrict__ b2,
                      const float* __restrict__ w1, const float* __restrict__ w2,
                      const float* __restrict__ wdw, float* __restrict__ hout) {
  __shared__ float z[8][256];
  __shared__ float red[256];
  __shared__ float zr[32][8];
  __shared__ float z1s[32];
  int n = blockIdx.x, t = threadIdx.x;
  int b = n >> 10, l = n & (L_ - 1);
  float xr[4];
  #pragma unroll
  for (int ci = 0; ci < 4; ++ci) {
    xr[ci] = x[((size_t)(b*C_+ci)*L_ + l)*F_ + t];
    z[ci][t]   = xr[ci];
    z[4+ci][t] = o[((size_t)n * C_ + ci) * F_ + t];
  }
  __syncthreads();
  #pragma unroll
  for (int c = 0; c < 8; ++c) {
    float vv = z[c][t];
    float s  = block_reduce_sum(vv, red);
    float m  = s * (1.f/256.f);
    float s2 = block_reduce_sum(vv*vv, red);
    float var = s2 * (1.f/256.f) - m*m;
    z[c][t] = (vv - m) * rsqrtf(var + EPS_) * g2[c] + b2[c];
  }
  __syncthreads();
  { // z1[c][g] = sum_f z[c][f] * w1[c,g,f]; 32 (c,g) pairs x 8 partials each
    int p = t >> 3, j = t & 7;
    int c = p >> 2, g = p & 3;
    const float* wp = w1 + (size_t)(c*EXP_ + g) * F_ + j*32;
    const float* zp = &z[c][j*32];
    float s = 0.f;
    #pragma unroll
    for (int i = 0; i < 32; ++i) s += zp[i] * wp[i];
    zr[p][j] = s;
  }
  __syncthreads();
  if (t < 32) {
    float s = 0.f;
    #pragma unroll
    for (int j = 0; j < 8; ++j) s += zr[t][j];
    z1s[t] = (s >= 0.f) ? s : 0.01f * s;   // leaky_relu
  }
  __syncthreads();
  float z2c[8];
  #pragma unroll
  for (int c = 0; c < 8; ++c) {
    float s = 0.f;
    #pragma unroll
    for (int g = 0; g < 4; ++g) s += z1s[c*4+g] * w2[((size_t)c*F_ + t)*EXP_ + g];
    z2c[c] = s;
  }
  #pragma unroll
  for (int d = 0; d < 4; ++d) {
    float s = 0.f;
    #pragma unroll
    for (int c = 0; c < 8; ++c) s += z2c[c] * wdw[d*8 + c];
    hout[((size_t)(b*C_+d)*L_ + l)*F_ + t] = xr[d] + s;
  }
}

// ---- launcher ----
extern "C" void kernel_launch(void* const* d_in, const int* in_sizes, int n_in,
                              void* d_out, int out_size, void* d_ws, size_t ws_size,
                              hipStream_t stream) {
  (void)in_sizes; (void)n_in; (void)out_size; (void)ws_size;
  const float* x       = (const float*)d_in[0];
  const float* prevqk  = (const float*)d_in[1];
  const float* g1      = (const float*)d_in[2];
  const float* b1      = (const float*)d_in[3];
  const float* wq_conv = (const float*)d_in[4];
  const float* wk_conv = (const float*)d_in[5];
  const float* wv_conv = (const float*)d_in[6];
  const float* wq_pw   = (const float*)d_in[7];
  const float* wk_pw   = (const float*)d_in[8];
  const float* wv_pw   = (const float*)d_in[9];
  const float* wo_pw   = (const float*)d_in[10];
  const float* g2      = (const float*)d_in[11];
  const float* b2      = (const float*)d_in[12];
  const float* w1_pw   = (const float*)d_in[13];
  const float* w2_pw   = (const float*)d_in[14];
  const float* w2_dw   = (const float*)d_in[15];

  float* hout  = (float*)d_out;                       // (B,C,L,CH,1)
  float* qkout = hout + (size_t)B_*C_*L_*F_;          // (B,C,H,L,L)

  char* ws = (char*)d_ws;
  size_t off = 0;
  auto alloc = [&](size_t bytes) -> void* {
    off = (off + 255) & ~(size_t)255;
    void* p = ws + off;
    off += bytes;
    return p;
  };
  const size_t NCF = (size_t)N_ * C_ * F_;     // 2,097,152
  const size_t WPW = (size_t)C_ * F_ * F_;     // 262,144
  const size_t HLD = (size_t)BCH_ * L_ * D_;   // 2,097,152

  bf16_t* cq  = (bf16_t*)alloc(NCF * 2);
  bf16_t* ck  = (bf16_t*)alloc(NCF * 2);
  bf16_t* cv  = (bf16_t*)alloc(NCF * 2);
  bf16_t* wqb = (bf16_t*)alloc(WPW * 2);
  bf16_t* wkb = (bf16_t*)alloc(WPW * 2);
  bf16_t* wvb = (bf16_t*)alloc(WPW * 2);
  bf16_t* wob = (bf16_t*)alloc(WPW * 2);
  bf16_t* qhb = (bf16_t*)alloc(HLD * 2);
  bf16_t* khb = (bf16_t*)alloc(HLD * 2);
  bf16_t* vtb = (bf16_t*)alloc(HLD * 2);
  float*  rmx = (float*)alloc((size_t)BCH_ * L_ * 4);
  float*  rsm = (float*)alloc((size_t)BCH_ * L_ * 4);
  bf16_t* afb = (bf16_t*)alloc(NCF * 2);
  float*  of  = (float*)alloc(NCF * 4);

  k_f32_to_bf16<<<(int)((WPW + 255) / 256), 256, 0, stream>>>(wq_pw, wqb, (int)WPW);
  k_f32_to_bf16<<<(int)((WPW + 255) / 256), 256, 0, stream>>>(wk_pw, wkb, (int)WPW);
  k_f32_to_bf16<<<(int)((WPW + 255) / 256), 256, 0, stream>>>(wv_pw, wvb, (int)WPW);
  k_f32_to_bf16<<<(int)((WPW + 255) / 256), 256, 0, stream>>>(wo_pw, wob, (int)WPW);

  k_prep<<<N_, 256, 0, stream>>>(x, g1, b1, wq_conv, wk_conv, wv_conv, cq, ck, cv);

  k_pw_gemm_rot<<<1024, 256, 0, stream>>>(cq, wqb, qhb);   // q: GEMM + rotary
  k_pw_gemm_rot<<<1024, 256, 0, stream>>>(ck, wkb, khb);   // k: GEMM + rotary
  k_pw_gemm_vT <<<1024, 256, 0, stream>>>(cv, wvb, vtb);   // v: GEMM + transpose

  k_qk_fused<<<512, 256, 0, stream>>>(qhb, khb, prevqk, qkout, rmx, rsm);
  k_av<<<1024, 256, 0, stream>>>(qkout, vtb, rmx, rsm, afb);

  k_pw_gemm<<<1024, 256, 0, stream>>>(afb, wob, of);       // o-projection

  k_mlp<<<N_, 256, 0, stream>>>(x, of, g2, b2, w1_pw, w2_pw, w2_dw, hout);
}
